// testMamba_867583394439
// MI455X (gfx1250) — matrix operand order repaired and checked by reading, not compiled, not run on hardware
//
#include <hip/hip_runtime.h>
#include <hip/hip_bf16.h>
#include <math.h>

// ---------------------------------------------------------------------------
// VMamba SS2D block on MI455X (gfx1250), wave32.
// Bandwidth-bound (~6.5 GFLOP over ~250MB of f32 traffic; fits in 192MB L2),
// so GEMMs use v_wmma_f32_16x16x32_f16 (f16 in / f32 acc) and everything else
// is lean VALU. Selective scan: 1 lane per (b,k,d) channel, 16-state in regs;
// B/C tiles (16 rows x 64 steps, row stride L) are DMA'd into LDS by the
// Tensor Data Mover (tensor_load_to_lds + s_wait_tensorcnt) once per chunk
// and broadcast to all 12 waves of the workgroup.
// ---------------------------------------------------------------------------

#define B_   2
#define H_   64
#define W_   64
#define D_   192
#define DS_  16
#define DTR_ 12
#define K_   4
#define DI_  384
#define L_   4096
#define BL_  (B_ * L_)        // 8192
#define C2_  (2 * DI_)        // 768
#define XC_  (DTR_ + 2 * DS_) // 44 rows of x_dbl

typedef __attribute__((ext_vector_type(16))) _Float16 v16h;
typedef __attribute__((ext_vector_type(8)))  float    v8f;
typedef __attribute__((ext_vector_type(4)))  unsigned int u32x4;
typedef __attribute__((ext_vector_type(8)))  int      i32x8;
typedef __attribute__((ext_vector_type(4)))  int      i32x4;

#if defined(__HIP_DEVICE_COMPILE__) && __has_builtin(__builtin_amdgcn_tensor_load_to_lds)
#define HAVE_TDM 1
#else
#define HAVE_TDM 0
#endif

// Workspace layout (float offsets)
#define OFF_XZ   ((size_t)0)                               // (BL, 768)
#define OFF_XCV  (OFF_XZ   + (size_t)BL_ * C2_)            // (B, DI, L)
#define OFF_XS   (OFF_XCV  + (size_t)B_ * DI_ * L_)        // (B*K, DI, L)
#define OFF_XDBL (OFF_XS   + (size_t)B_ * K_ * DI_ * L_)   // (B*K, 44, L)
#define OFF_DTS  (OFF_XDBL + (size_t)B_ * K_ * XC_ * L_)   // (B*K, DI, L)
#define OFF_OUTY (OFF_DTS  + (size_t)B_ * K_ * DI_ * L_)   // (B*K, DI, L)
#define OFF_YG   (OFF_OUTY + (size_t)B_ * K_ * DI_ * L_)   // (BL, DI)

// ---------------------------------------------------------------------------
// Kernel 1: in_proj GEMM  xz = x @ W^T + b   (M=8192, N=768, K=192), WMMA f16
// ---------------------------------------------------------------------------
__global__ void __launch_bounds__(256) inproj_gemm(
    const float* __restrict__ X,    // (8192, 192)
    const float* __restrict__ Wt,   // (768, 192) row-major (in_proj_w)
    const float* __restrict__ bias, // (768)
    float* __restrict__ XZ)         // (8192, 768)
{
  const int lane = threadIdx.x & 31;
  const int wave = blockIdx.x * (blockDim.x >> 5) + (threadIdx.x >> 5);
  const int NT = C2_ / 16;                 // 48
  const int mt = wave / NT, nt = wave % NT;
  if (mt >= BL_ / 16) return;              // wave-uniform
  const int m0 = mt * 16, n0 = nt * 16;
  const int hi   = lane >> 4;              // half-wave selects K-halves
  const int mrow = m0 + (lane & 15);
  const int ncol = n0 + (lane & 15);
  v8f acc = {};
  for (int k0 = 0; k0 < D_; k0 += 32) {
    if (k0 + 32 < D_) {
      __builtin_prefetch(&X[(size_t)mrow * D_ + k0 + 32], 0, 1);
      __builtin_prefetch(&Wt[(size_t)ncol * D_ + k0 + 32], 0, 1);
    }
    v16h a, b;
    #pragma unroll
    for (int e = 0; e < 16; ++e) {         // A 16x32 f16 per ISA layout
      int p = e >> 1;
      int kk = (p < 4) ? ((hi ? 8 : 0) + 2 * p + (e & 1))
                       : ((hi ? 24 : 16) + 2 * (p - 4) + (e & 1));
      a[e] = (_Float16)X[(size_t)mrow * D_ + k0 + kk];
    }
    const int kb = k0 + (hi ? 8 : 0);      // B 32x16: column per lane
    #pragma unroll
    for (int e = 0; e < 16; ++e)
      b[e] = (_Float16)Wt[(size_t)ncol * D_ + kb + e + ((e >> 3) << 3)];
    acc = __builtin_amdgcn_wmma_f32_16x16x32_f16(
        false, a, false, b, (short)0, acc, false, false);
  }
  const float bv = bias[ncol];
  #pragma unroll
  for (int r = 0; r < 8; ++r)              // D: M = r + 8*hi, N = lane&15
    XZ[(size_t)(m0 + 8 * hi + r) * C2_ + ncol] = acc[r] + bv;
}

// ---------------------------------------------------------------------------
// Kernel 2: depthwise 3x3 conv + bias + SiLU  -> xc (B, DI, L)
// ---------------------------------------------------------------------------
__global__ void conv_silu(const float* __restrict__ XZ,
                          const float* __restrict__ CW,  // (DI, 9)
                          const float* __restrict__ CB,  // (DI)
                          float* __restrict__ XCv)       // (B, DI, L)
{
  int idx = blockIdx.x * blockDim.x + threadIdx.x;
  if (idx >= B_ * DI_ * L_) return;
  const int l = idx & (L_ - 1);
  const int d = (idx >> 12) % DI_;
  const int b = idx / (DI_ * L_);
  const int h = l >> 6, w = l & 63;
  float s = CB[d];
  #pragma unroll
  for (int ky = -1; ky <= 1; ++ky) {
    const int hh = h + ky;
    if (hh < 0 || hh > H_ - 1) continue;
    #pragma unroll
    for (int kx = -1; kx <= 1; ++kx) {
      const int ww = w + kx;
      if (ww < 0 || ww > W_ - 1) continue;
      s += XZ[((size_t)b * L_ + hh * W_ + ww) * C2_ + d] *
           CW[d * 9 + (ky + 1) * 3 + (kx + 1)];
    }
  }
  XCv[idx] = s / (1.f + expf(-s));  // SiLU
}

// ---------------------------------------------------------------------------
// Kernel 3: build 4 directional scans xs (B*K, DI, L)
// ---------------------------------------------------------------------------
__global__ void build_xs(const float* __restrict__ XCv, float* __restrict__ XS)
{
  int idx = blockIdx.x * blockDim.x + threadIdx.x;
  if (idx >= B_ * DI_ * L_) return;
  const int l = idx & (L_ - 1);
  const int d = (idx >> 12) % DI_;
  const int b = idx / (DI_ * L_);
  const int lt = (l & 63) * 64 + (l >> 6);       // HW <-> WH transpose
  const float* src = XCv + ((size_t)b * DI_ + d) * L_;
  const float v0 = src[l];
  const float v1 = src[lt];
  const size_t kstride = (size_t)DI_ * L_;
  float* dst = XS + ((size_t)(b * K_) * DI_ + d) * L_;
  dst[l] = v0;                                   // k=0: hw order
  dst[kstride + l] = v1;                         // k=1: wh order
  dst[2 * kstride + (L_ - 1 - l)] = v0;          // k=2: hw reversed
  dst[3 * kstride + (L_ - 1 - l)] = v1;          // k=3: wh reversed
}

// ---------------------------------------------------------------------------
// Kernel 4: x_dbl = x_proj_w[k] @ xs[b,k]  (M=44->48, N=4096, K=384), WMMA
// ---------------------------------------------------------------------------
__global__ void __launch_bounds__(256) xdbl_gemm(
    const float* __restrict__ XPW,  // (K, 44, 384)
    const float* __restrict__ XS,   // (B*K, 384, 4096)
    float* __restrict__ XDBL)       // (B*K, 44, 4096)
{
  const int lane = threadIdx.x & 31;
  const int wave = blockIdx.x * (blockDim.x >> 5) + (threadIdx.x >> 5);
  const int tilesPerBK = 3 * (L_ / 16);          // 3 x 256
  const int bk = wave / tilesPerBK;
  if (bk >= B_ * K_) return;                     // wave-uniform
  const int rem = wave % tilesPerBK;
  const int mt = rem >> 8, nt = rem & 255;
  const int m0 = mt * 16, n0 = nt * 16;
  const float* A  = XPW + (size_t)(bk & 3) * XC_ * DI_;
  const float* Bm = XS + (size_t)bk * DI_ * L_;
  const int hi   = lane >> 4;
  const int mrow = m0 + (lane & 15);
  const int ncol = n0 + (lane & 15);
  v8f acc = {};
  for (int k0 = 0; k0 < DI_; k0 += 32) {
    v16h a, b;
    #pragma unroll
    for (int e = 0; e < 16; ++e) {
      int p = e >> 1;
      int kk = (p < 4) ? ((hi ? 8 : 0) + 2 * p + (e & 1))
                       : ((hi ? 24 : 16) + 2 * (p - 4) + (e & 1));
      float av = (mrow < XC_) ? A[(size_t)mrow * DI_ + k0 + kk] : 0.f;
      a[e] = (_Float16)av;
    }
    const int kb = k0 + (hi ? 8 : 0);
    #pragma unroll
    for (int e = 0; e < 16; ++e)
      b[e] = (_Float16)Bm[(size_t)(kb + e + ((e >> 3) << 3)) * L_ + ncol];
    acc = __builtin_amdgcn_wmma_f32_16x16x32_f16(
        false, a, false, b, (short)0, acc, false, false);
  }
  #pragma unroll
  for (int r = 0; r < 8; ++r) {
    const int m = m0 + 8 * hi + r;
    if (m < XC_)
      XDBL[((size_t)bk * XC_ + m) * L_ + ncol] = acc[r];
  }
}

// ---------------------------------------------------------------------------
// Kernel 5: dts = softplus(dt_projs_w @ dts_raw + b)  (K-dim = 12 -> VALU)
// ---------------------------------------------------------------------------
__global__ void dtproj_softplus(const float* __restrict__ XDBL,
                                const float* __restrict__ DTW, // (K, DI, 12)
                                const float* __restrict__ DTB, // (K, DI)
                                float* __restrict__ DTS)       // (B*K, DI, L)
{
  int idx = blockIdx.x * blockDim.x + threadIdx.x;
  if (idx >= B_ * K_ * DI_ * L_) return;
  const int l = idx & (L_ - 1);
  const int d = (idx >> 12) % DI_;
  const int bk = idx / (DI_ * L_);
  const int k = bk & 3;
  const float* xr = XDBL + (size_t)bk * XC_ * L_ + l;    // rows 0..11
  const float* wr = DTW + ((size_t)k * DI_ + d) * DTR_;
  float s = DTB[k * DI_ + d];
  #pragma unroll
  for (int r = 0; r < DTR_; ++r) s += xr[(size_t)r * L_] * wr[r];
  DTS[idx] = (s > 20.f) ? s : log1pf(expf(s));           // softplus
}

// ---------------------------------------------------------------------------
// TDM helper: DMA a 16-row x 64-col f32 tile (row stride 4096 elements) from
// global into LDS at ldsOff, row-major. D# packed per CDNA5 ISA 8.3/8.4.
// ---------------------------------------------------------------------------
#if HAVE_TDM
__device__ __forceinline__ void tdm_load_tile16x64(const float* gsrc,
                                                   unsigned ldsOff)
{
  const unsigned long long ga = (unsigned long long)(uintptr_t)gsrc;
  u32x4 g0;
  g0.x = 1u;                                        // count=1, user mode
  g0.y = ldsOff;                                    // lds_addr [63:32]
  g0.z = (unsigned)ga;                              // global_addr[31:0]
  g0.w = (unsigned)((ga >> 32) & 0x01FFFFFFu)       // global_addr[56:32]
       | 0x80000000u;                               // type=2 ("image")
  i32x8 g1;
  g1[0] = 0x00020000;                               // data_size=4B, no mcast
  g1[1] = (int)((unsigned)(L_ & 0xFFFF) << 16);     // tensor_dim0 = 4096
  g1[2] = (int)((unsigned)DS_ << 16);               // tensor_dim1 = 16
  g1[3] = (int)(64u << 16);                         // tile_dim0  = 64
  g1[4] = DS_;                                      // tile_dim1  = 16
  g1[5] = L_;                                       // tensor_dim0_stride
  g1[6] = 0;
  g1[7] = 0;
  i32x4 z4 = {0, 0, 0, 0};
#if __clang_major__ >= 23
  i32x8 z8 = {0, 0, 0, 0, 0, 0, 0, 0};
  __builtin_amdgcn_tensor_load_to_lds(g0, g1, z4, z4, z8, 0);
#else
  __builtin_amdgcn_tensor_load_to_lds(g0, g1, z4, z4, 0);
#endif
}
#endif

// ---------------------------------------------------------------------------
// Kernel 6: selective scan. 1 block per (b,k); lane = channel d; 16-state regs
// B/C chunk tiles staged via TDM (wave 0 issues, s_wait_tensorcnt, barrier).
// ---------------------------------------------------------------------------
__global__ void __launch_bounds__(DI_) selective_scan(
    const float* __restrict__ XS, const float* __restrict__ DTS,
    const float* __restrict__ XDBL, const float* __restrict__ ALOG,
    const float* __restrict__ DsP, float* __restrict__ OUTY)
{
  const int bk = blockIdx.x;                // b*K + k
  const int k  = bk & 3;
  const int d  = threadIdx.x;               // 0..383
  __shared__ float Bt[DS_][64];             // row-major: matches TDM layout
  __shared__ float Ct[DS_][64];
  float Aa[DS_], h[DS_];
  #pragma unroll
  for (int n = 0; n < DS_; ++n) {
    Aa[n] = -expf(ALOG[((size_t)k * DI_ + d) * DS_ + n]);
    h[n] = 0.f;
  }
  const float Dv = DsP[k * DI_ + d];
  const float* u_p  = XS  + ((size_t)bk * DI_ + d) * L_;
  const float* dt_p = DTS + ((size_t)bk * DI_ + d) * L_;
  const float* Bg = XDBL + ((size_t)bk * XC_ + DTR_) * L_;
  const float* Cg = Bg + (size_t)DS_ * L_;
  float* y_p = OUTY + ((size_t)bk * DI_ + d) * L_;

  for (int l0 = 0; l0 < L_; l0 += 64) {
    __syncthreads();                        // LDS free before refill
#if HAVE_TDM
    if (threadIdx.x < 32) {                 // wave 0 drives the TDM
      tdm_load_tile16x64(Bg + l0, (unsigned)(uintptr_t)&Bt[0][0]);
      tdm_load_tile16x64(Cg + l0, (unsigned)(uintptr_t)&Ct[0][0]);
      __builtin_amdgcn_s_wait_tensorcnt(0);
    }
#else
    for (int t = threadIdx.x; t < 64 * DS_; t += DI_) {
      const int n = t >> 6, i = t & 63;
      Bt[n][i] = Bg[(size_t)n * L_ + l0 + i];
      Ct[n][i] = Cg[(size_t)n * L_ + l0 + i];
    }
#endif
    __syncthreads();                        // publish tile to all 12 waves
    for (int i = 0; i < 64; ++i) {
      const float u  = u_p[l0 + i];
      const float dt = dt_p[l0 + i];
      float y = 0.f;
      #pragma unroll
      for (int n = 0; n < DS_; ++n) {
        const float dA = expf(dt * Aa[n]);
        h[n] = h[n] * dA + dt * Bt[n][i] * u;
        y += h[n] * Ct[n][i];
      }
      y_p[l0 + i] = y + Dv * u;
    }
  }
}

// ---------------------------------------------------------------------------
// Kernel 7: merge 4 directions + LayerNorm + SiLU gate. 1 wave per (b,l) row.
// ---------------------------------------------------------------------------
__global__ void __launch_bounds__(256) merge_ln_gate(
    const float* __restrict__ OUTY, const float* __restrict__ XZ,
    const float* __restrict__ LNW, const float* __restrict__ LNB,
    float* __restrict__ YG)         // (BL, DI)
{
  const int lane = threadIdx.x & 31;
  const int row  = blockIdx.x * (blockDim.x >> 5) + (threadIdx.x >> 5);
  if (row >= BL_) return;
  const int b = row >> 12;
  const int l = row & (L_ - 1);
  const int lt = (l & 63) * 64 + (l >> 6);
  const size_t cs = (size_t)DI_ * L_;              // per-direction stride
  const float* y0 = OUTY + (size_t)(b * K_ + 0) * cs;
  const float* y1 = OUTY + (size_t)(b * K_ + 1) * cs;
  const float* y2 = OUTY + (size_t)(b * K_ + 2) * cs;
  const float* y3 = OUTY + (size_t)(b * K_ + 3) * cs;
  float vals[12];
  float sum = 0.f, sumsq = 0.f;
  #pragma unroll
  for (int j = 0; j < 12; ++j) {
    const int d = lane + j * 32;
    const float v = y0[(size_t)d * L_ + l]
                  + y2[(size_t)d * L_ + (L_ - 1 - l)]
                  + y1[(size_t)d * L_ + lt]
                  + y3[(size_t)d * L_ + (L_ - 1 - lt)];
    vals[j] = v; sum += v; sumsq += v * v;
  }
  #pragma unroll
  for (int off = 16; off >= 1; off >>= 1) {        // wave32 reduction
    sum   += __shfl_xor(sum, off, 32);
    sumsq += __shfl_xor(sumsq, off, 32);
  }
  const float mu  = sum / (float)DI_;
  const float var = sumsq / (float)DI_ - mu * mu;
  const float rstd = rsqrtf(var + 1e-5f);
  #pragma unroll
  for (int j = 0; j < 12; ++j) {
    const int d = lane + j * 32;
    const float y = (vals[j] - mu) * rstd * LNW[d] + LNB[d];
    const float z = XZ[(size_t)row * C2_ + DI_ + d];
    YG[(size_t)row * DI_ + d] = y * (z / (1.f + expf(-z)));
  }
}

// ---------------------------------------------------------------------------
// Kernel 8: out_proj GEMM  out = yg @ out_proj_w^T (M=8192, N=192, K=384)
// ---------------------------------------------------------------------------
__global__ void __launch_bounds__(256) outproj_gemm(
    const float* __restrict__ YG,   // (8192, 384)
    const float* __restrict__ OW,   // (192, 384) row-major
    float* __restrict__ OUT)        // (8192, 192)
{
  const int lane = threadIdx.x & 31;
  const int wave = blockIdx.x * (blockDim.x >> 5) + (threadIdx.x >> 5);
  const int NT = D_ / 16;                   // 12
  const int mt = wave / NT, nt = wave % NT;
  if (mt >= BL_ / 16) return;               // wave-uniform
  const int m0 = mt * 16, n0 = nt * 16;
  const int hi   = lane >> 4;
  const int mrow = m0 + (lane & 15);
  const int ncol = n0 + (lane & 15);
  v8f acc = {};
  for (int k0 = 0; k0 < DI_; k0 += 32) {
    if (k0 + 32 < DI_) {
      __builtin_prefetch(&YG[(size_t)mrow * DI_ + k0 + 32], 0, 1);
      __builtin_prefetch(&OW[(size_t)ncol * DI_ + k0 + 32], 0, 1);
    }
    v16h a, b;
    #pragma unroll
    for (int e = 0; e < 16; ++e) {
      int p = e >> 1;
      int kk = (p < 4) ? ((hi ? 8 : 0) + 2 * p + (e & 1))
                       : ((hi ? 24 : 16) + 2 * (p - 4) + (e & 1));
      a[e] = (_Float16)YG[(size_t)mrow * DI_ + k0 + kk];
    }
    const int kb = k0 + (hi ? 8 : 0);
    #pragma unroll
    for (int e = 0; e < 16; ++e)
      b[e] = (_Float16)OW[(size_t)ncol * DI_ + kb + e + ((e >> 3) << 3)];
    acc = __builtin_amdgcn_wmma_f32_16x16x32_f16(
        false, a, false, b, (short)0, acc, false, false);
  }
  #pragma unroll
  for (int r = 0; r < 8; ++r)
    OUT[(size_t)(m0 + 8 * hi + r) * D_ + ncol] = acc[r];
}

// ---------------------------------------------------------------------------
extern "C" void kernel_launch(void* const* d_in, const int* in_sizes, int n_in,
                              void* d_out, int out_size, void* d_ws, size_t ws_size,
                              hipStream_t stream) {
  const float* x       = (const float*)d_in[0];
  const float* ipw     = (const float*)d_in[1];
  const float* ipb     = (const float*)d_in[2];
  const float* convw   = (const float*)d_in[3];
  const float* convb   = (const float*)d_in[4];
  const float* xpw     = (const float*)d_in[5];
  const float* dtw     = (const float*)d_in[6];
  const float* dtb     = (const float*)d_in[7];
  const float* alogs   = (const float*)d_in[8];
  const float* dsv     = (const float*)d_in[9];
  const float* lnw     = (const float*)d_in[10];
  const float* lnb     = (const float*)d_in[11];
  const float* opw     = (const float*)d_in[12];
  float* out = (float*)d_out;
  float* ws  = (float*)d_ws;

  float* xz   = ws + OFF_XZ;
  float* xcv  = ws + OFF_XCV;
  float* xs   = ws + OFF_XS;
  float* xdbl = ws + OFF_XDBL;
  float* dts  = ws + OFF_DTS;
  float* outy = ws + OFF_OUTY;
  float* yg   = ws + OFF_YG;

  // 1) in_proj: 512x48 = 24576 wave-tiles, 8 waves/block
  inproj_gemm<<<24576 / 8, 256, 0, stream>>>(x, ipw, ipb, xz);
  // 2) depthwise conv + SiLU
  conv_silu<<<(B_ * DI_ * L_ + 255) / 256, 256, 0, stream>>>(xz, convw, convb, xcv);
  // 3) 4 directional scans
  build_xs<<<(B_ * DI_ * L_ + 255) / 256, 256, 0, stream>>>(xcv, xs);
  // 4) x_dbl: 8 * 3 * 256 = 6144 wave-tiles
  xdbl_gemm<<<6144 / 8, 256, 0, stream>>>(xpw, xs, xdbl);
  // 5) dt projection + softplus
  dtproj_softplus<<<(B_ * K_ * DI_ * L_ + 255) / 256, 256, 0, stream>>>(xdbl, dtw, dtb, dts);
  // 6) selective scan: one block per (b,k), TDM-staged B/C tiles
  selective_scan<<<B_ * K_, DI_, 0, stream>>>(xs, dts, xdbl, alogs, dsv, outy);
  // 7) merge + LN + gate: 8192 rows, 8 waves/block
  merge_ln_gate<<<BL_ / 8, 256, 0, stream>>>(outy, xz, lnw, lnb, yg);
  // 8) out_proj: 512x12 = 6144 wave-tiles
  outproj_gemm<<<6144 / 8, 256, 0, stream>>>(yg, opw, out);
}
